// OwnVanillaRNN_18708877541421
// MI455X (gfx1250) — matrix operand order repaired and checked by reading, not compiled, not run on hardware
//
#include <hip/hip_runtime.h>
#include <hip/hip_bf16.h>

// Vanilla RNN scan, MI455X / gfx1250.
// y = scan(tanh(x@Wx^T + bx + h@Wh^T + bh)) @ Wy^T + by
// 16 WGs (16 batch rows each) x 8 wave32; per-wave 64 hidden cols.
// bf16 WMMA (v_wmma_f32_16x16x32_bf16) with f32 accumulate; h double-buffered
// in LDS; Wh streamed from L2 (bf16 copy in workspace); hardware v_tanh_f32.

#define RNN_INPUT   64
#define RNN_HIDDEN  512
#define RNN_OUTPUT  64
#define RNN_BATCH   256
#define RNN_SEQ     512
#define LDS_STRIDE  520   // 512 + 8 bf16 pad -> 4-bank shift per row, conflict-free b128 LDS loads

typedef __attribute__((ext_vector_type(16))) __bf16 v16bf;
typedef __attribute__((ext_vector_type(8)))  __bf16 v8bf;
typedef __attribute__((ext_vector_type(8)))  float  v8f;

#if defined(__HIP_DEVICE_COMPILE__) && !__has_builtin(__builtin_amdgcn_wmma_f32_16x16x32_bf16)
#error "gfx1250 bf16 WMMA builtin not available"
#endif

// CDNA5 has a hardware V_TANH_F32 (TRANS unit).
__device__ __forceinline__ float fast_tanh(float v) {
#if __has_builtin(__builtin_amdgcn_tanhf)
    return __builtin_amdgcn_tanhf(v);
#elif __has_builtin(__builtin_amdgcn_tanh_f32)
    return __builtin_amdgcn_tanh_f32(v);
#else
    return tanhf(v);
#endif
}

__global__ __launch_bounds__(256) void rnn_cvt_weights(
    const float* __restrict__ Wx, const float* __restrict__ Wh,
    const float* __restrict__ Wy,
    __bf16* __restrict__ Whb, __bf16* __restrict__ Wxb, __bf16* __restrict__ Wyb)
{
    int idx = blockIdx.x * 256 + threadIdx.x;
    if (idx < RNN_HIDDEN * RNN_HIDDEN) Whb[idx] = (__bf16)Wh[idx];
    if (idx < RNN_HIDDEN * RNN_INPUT) {
        Wxb[idx] = (__bf16)Wx[idx];
        Wyb[idx] = (__bf16)Wy[idx];   // Wy is OUTPUT x HIDDEN = same 32768 elements
    }
}

__global__ __launch_bounds__(256) void rnn_scan(
    const float*  __restrict__ x,   const float* __restrict__ bx,
    const float*  __restrict__ bh,  const float* __restrict__ by,
    const __bf16* __restrict__ Whb, const __bf16* __restrict__ Wxb,
    const __bf16* __restrict__ Wyb, float* __restrict__ out)
{
    __shared__ __bf16 hbuf[2][16][LDS_STRIDE];

    const int tid   = threadIdx.x;
    const int wave  = tid >> 5;
    const int lane  = tid & 31;
    const int half  = lane >> 4;     // which 16-lane half of the wave
    const int lr    = lane & 15;
    const int wbase = wave * 64;     // this wave's hidden-column base
    const int bb    = blockIdx.x * 16;

    // h0 = 0 (zero both LDS buffers cooperatively)
    for (int i = tid; i < 2 * 16 * LDS_STRIDE; i += 256)
        ((__bf16*)hbuf)[i] = (__bf16)0.0f;

    // Per-step accumulator init: bx[col] + bh[col] (depends on N only).
    float binit[4];
#pragma unroll
    for (int j = 0; j < 4; ++j) {
        int col = wbase + j * 16 + lr;
        binit[j] = bx[col] + bh[col];
    }

    // Loop-invariant B fragments of Wx^T: B[k,n] = Wx[n,k].
    // lane (n=lr) reads 16 consecutive K from row (wbase+j*16+lr) of Wx.
    v16bf bwx[4][2];
#pragma unroll
    for (int j = 0; j < 4; ++j)
#pragma unroll
        for (int kf = 0; kf < 2; ++kf)
            bwx[j][kf] = *(const v16bf*)(Wxb + (wbase + j * 16 + lr) * RNN_INPUT
                                             + kf * 32 + half * 16);

    __syncthreads();

    const float* xrow = x + (size_t)(bb + lr) * RNN_SEQ * RNN_INPUT;

    for (int t = 0; t < RNN_SEQ; ++t) {
        const int cur = t & 1, nxt = cur ^ 1;

        v8f acc[4];
#pragma unroll
        for (int j = 0; j < 4; ++j)
#pragma unroll
            for (int g = 0; g < 8; ++g) acc[j][g] = binit[j];

        // ---- fused input projection: A = x[b_tile, t, :] (16x64 f32 -> bf16)
        const float* xp = xrow + (size_t)t * RNN_INPUT;
        if (t + 1 < RNN_SEQ)
            __builtin_prefetch(xp + RNN_INPUT, 0, 1);  // next step's x row
        v16bf ax[2];
#pragma unroll
        for (int kf = 0; kf < 2; ++kf) {
            const float* p0 = xp + kf * 32 + half * 16;  // K kb..kb+7 (lane<16) / kb+16..23
            const float* p1 = p0 + 8;                    // K kb+8..15 / kb+24..31
            v16bf a;
#pragma unroll
            for (int i = 0; i < 8; ++i) a[i]     = (__bf16)p0[i];
#pragma unroll
            for (int i = 0; i < 8; ++i) a[8 + i] = (__bf16)p1[i];
            ax[kf] = a;
        }
#pragma unroll
        for (int j = 0; j < 4; ++j)
#pragma unroll
            for (int kf = 0; kf < 2; ++kf)
                acc[j] = __builtin_amdgcn_wmma_f32_16x16x32_bf16(
                    false, ax[kf], false, bwx[j][kf], (short)0, acc[j], false, false);

        // ---- recurrence: h_{t-1} @ Wh^T, K = 512 in 16 fragments of 32,
        // B fragments streamed from the L2-resident bf16 Wh copy.
        const __bf16* arow = &hbuf[cur][lr][0];
#pragma unroll 4
        for (int kt = 0; kt < 16; ++kt) {
            const int kb = kt * 32;
            v8bf lo = *(const v8bf*)(arow + kb + half * 16);      // ds_load_b128
            v8bf hi = *(const v8bf*)(arow + kb + half * 16 + 8);  // ds_load_b128
            v16bf a = __builtin_shufflevector(lo, hi,
                0,1,2,3,4,5,6,7,8,9,10,11,12,13,14,15);
#pragma unroll
            for (int j = 0; j < 4; ++j) {
                // B[k,n] = Wh[n,k]: contiguous 32B from row (wbase+j*16+lr) of Wh
                v16bf b = *(const v16bf*)(Whb + (size_t)(wbase + j * 16 + lr) * RNN_HIDDEN
                                               + kb + half * 16);
                acc[j] = __builtin_amdgcn_wmma_f32_16x16x32_bf16(
                    false, a, false, b, (short)0, acc[j], false, false);
            }
        }

        // ---- tanh (hardware V_TANH_F32) and write h_t (C-layout scatter)
#pragma unroll
        for (int j = 0; j < 4; ++j)
#pragma unroll
            for (int g = 0; g < 8; ++g)
                hbuf[nxt][half * 8 + g][wbase + j * 16 + lr] =
                    (__bf16)fast_tanh(acc[j][g]);

        __syncthreads();  // single barrier/step is safe with double buffering
    }

    // ---- output projection: y = hT @ Wy^T + by.  hT sits in hbuf[0]
    // (last write was step t=511 -> nxt=0). Waves 0..3 each do one 16-col N-tile.
    if (wave < 4) {
        const int nb = wave * 16;
        v8f acc;
        float bi = by[nb + lr];
#pragma unroll
        for (int g = 0; g < 8; ++g) acc[g] = bi;

        const __bf16* arow = &hbuf[0][lr][0];
#pragma unroll 4
        for (int kt = 0; kt < 16; ++kt) {
            const int kb = kt * 32;
            v8bf lo = *(const v8bf*)(arow + kb + half * 16);
            v8bf hi = *(const v8bf*)(arow + kb + half * 16 + 8);
            v16bf a = __builtin_shufflevector(lo, hi,
                0,1,2,3,4,5,6,7,8,9,10,11,12,13,14,15);
            v16bf b = *(const v16bf*)(Wyb + (size_t)(nb + lr) * RNN_HIDDEN
                                           + kb + half * 16);
            acc = __builtin_amdgcn_wmma_f32_16x16x32_bf16(
                false, a, false, b, (short)0, acc, false, false);
        }
#pragma unroll
        for (int g = 0; g < 8; ++g)
            out[(size_t)(bb + half * 8 + g) * RNN_OUTPUT + nb + lr] = acc[g];
    }
}

extern "C" void kernel_launch(void* const* d_in, const int* in_sizes, int n_in,
                              void* d_out, int out_size, void* d_ws, size_t ws_size,
                              hipStream_t stream) {
    const float* x  = (const float*)d_in[0];
    const float* Wx = (const float*)d_in[1];
    const float* bx = (const float*)d_in[2];
    const float* Wh = (const float*)d_in[3];
    const float* bh = (const float*)d_in[4];
    const float* Wy = (const float*)d_in[5];
    const float* by = (const float*)d_in[6];
    float* out = (float*)d_out;

    // Workspace: bf16 copies of the weights (640 KB total).
    __bf16* Whb = (__bf16*)d_ws;                          // 512*512
    __bf16* Wxb = Whb + RNN_HIDDEN * RNN_HIDDEN;          // 512*64
    __bf16* Wyb = Wxb + RNN_HIDDEN * RNN_INPUT;           // 64*512

    rnn_cvt_weights<<<(RNN_HIDDEN * RNN_HIDDEN) / 256, 256, 0, stream>>>(
        Wx, Wh, Wy, Whb, Wxb, Wyb);
    rnn_scan<<<RNN_BATCH / 16, 256, 0, stream>>>(
        x, bx, bh, by, Whb, Wxb, Wyb, out);
}
